// LoadBalancedRouter_50697793962042
// MI455X (gfx1250) — matrix operand order repaired and checked by reading, not compiled, not run on hardware
//
#include <hip/hip_runtime.h>
#include <hip/hip_bf16.h>

typedef __attribute__((ext_vector_type(16))) __bf16 v16bf;
typedef __attribute__((ext_vector_type(8)))  __bf16 v8bf;
typedef __attribute__((ext_vector_type(8)))  float  v8f;
typedef __attribute__((ext_vector_type(4)))  float  v4f;
typedef unsigned int u32x4 __attribute__((ext_vector_type(4)));
typedef int          i32x4 __attribute__((ext_vector_type(4)));
typedef int          i32x8 __attribute__((ext_vector_type(8)));

#define D_MODEL 2048
#define N_EXP   64
#define KSTEP   32
#define NKS     (D_MODEL / KSTEP)   // 64 K-steps
#define BROW    40                  // padded LDS row pitch in bf16 (80 B) -> bank-conflict-free

// ---------------------------------------------------------------------------
// Kernel 1: split W (fp32) into bf16 hi/lo planes (contiguous 128x2048 bf16
// "Wsplit" tensor: rows 0..63 = hi, rows 64..127 = lo); zero expert accums.
// ---------------------------------------------------------------------------
__global__ void router_prep(const float* __restrict__ W,
                            __bf16* __restrict__ Whi,
                            __bf16* __restrict__ Wlo,
                            float* __restrict__ gExpert) {
  int i = blockIdx.x * blockDim.x + threadIdx.x;
  if (i < N_EXP * D_MODEL) {
    float f = W[i];
    __bf16 h = (__bf16)f;
    Whi[i] = h;
    Wlo[i] = (__bf16)(f - (float)h);
  }
  if (blockIdx.x == 0 && threadIdx.x < N_EXP) gExpert[threadIdx.x] = 0.0f;
}

// ---------------------------------------------------------------------------
// TDM stage: copy a 32-col x 128-row bf16 strip of Wsplit into LDS.
// D# per CDNA5 ISA ch.8: 2D tensor 2048x128 (stride 2048), tile 32x128,
// data_size=2B, LDS pad 4 DWORDs after every 16 DWORDs (64B row -> 80B pitch).
// ---------------------------------------------------------------------------
__device__ __forceinline__ void tdm_stage_B(const __bf16* gsrc, __bf16* ldsDst,
                                            int lane) {
#if __has_builtin(__builtin_amdgcn_tensor_load_to_lds)
  unsigned long long ga = (unsigned long long)(size_t)gsrc;
  unsigned ldsAddr = (unsigned)(size_t)ldsDst;   // low 32 bits of flat ptr = LDS offset
  u32x4 g0;
  g0[0] = 1u;                                        // count=1, user descriptor
  g0[1] = ldsAddr;                                   // lds_addr
  g0[2] = (unsigned)ga;                              // global_addr[31:0]
  g0[3] = ((unsigned)(ga >> 32) & 0x01FFFFFFu)       // global_addr[56:32]
          | (2u << 30);                              // type = 2 ("image")
  i32x8 g1;
  g1[0] = 0x00010000                                 // data_size = 2 bytes
          | (1 << 20)                                // pad_enable
          | (3 << 22)                                // pad_interval: 16 DWORDs
          | (3 << 25);                               // pad_amount: 4 DWORDs (16 B)
  g1[1] = (int)(2048u << 16);                        // tensor_dim0 = 2048
  g1[2] = (int)(128u  << 16);                        // tensor_dim1 = 128
  g1[3] = (int)(32u   << 16);                        // tile_dim0 = 32
  g1[4] = 128;                                       // tile_dim1 = 128, tile_dim2 = 0
  g1[5] = 2048;                                      // tensor_dim0_stride = 2048
  g1[6] = 0;
  g1[7] = 0;
  i32x4 gz = {0, 0, 0, 0};
#if defined(__clang_major__) && __clang_major__ >= 23
  i32x8 gz8 = {0, 0, 0, 0, 0, 0, 0, 0};
  __builtin_amdgcn_tensor_load_to_lds(g0, g1, gz, gz, gz8, 0);
#else
  __builtin_amdgcn_tensor_load_to_lds(g0, g1, gz, gz, 0);
#endif
  (void)lane;
#else
  // Fallback (should not trigger on probed toolchains): wave-cooperative copy.
  for (int r = lane; r < 128; r += 32) {
    const v8bf* s = (const v8bf*)(gsrc + (size_t)r * D_MODEL);
    v8bf* d = (v8bf*)(ldsDst + r * BROW);
    d[0] = s[0];
    d[1] = s[1];
  }
#endif
}

// ---------------------------------------------------------------------------
// Kernel 2: WMMA GEMM (split-bf16 emulated fp32) + fused router epilogue.
// Block = 128 threads = 4 waves; each wave: 16 token rows x 64 experts.
// B (shared by all 4 waves) is TDM-staged into double-buffered LDS.
// ---------------------------------------------------------------------------
__global__ __launch_bounds__(128) void router_gemm(
    const float* __restrict__ x,
    const __bf16* __restrict__ Wsplit,   // [128][2048] bf16: hi rows then lo rows
    float* __restrict__ gExpert,
    float* __restrict__ out,
    int tokens) {
  __shared__ __bf16 ldsB[2][128 * BROW];   // double-buffered B strip (2 x 10 KB)
  __shared__ float  lgs[4][16][N_EXP];     // per-wave logits tile
  __shared__ float  wgExpert[N_EXP];       // block-level expert prob sums

  const int tid  = threadIdx.x;
  const int wv   = tid >> 5;
  const int lane = tid & 31;
  const int lm   = lane & 15;
  const int lh   = lane >> 4;

  if (tid < N_EXP) wgExpert[tid] = 0.0f;

  const int rowBase = blockIdx.x * 64 + wv * 16;
  const int m = rowBase + lm;              // A-matrix row owned by this lane
  const float* xrow = x + (size_t)m * D_MODEL;

  v8f zero = {};
  v8f acc[4];
#pragma unroll
  for (int t = 0; t < 4; ++t) acc[t] = zero;

  // Prologue: stage K-step 0 into buffer 0.
  if (wv == 0) tdm_stage_B(Wsplit, &ldsB[0][0], lane);

  for (int k = 0; k < NKS; ++k) {
    const int cur = k & 1;
    const int nxt = cur ^ 1;
    const int K = k * KSTEP;

    // --- A tile (issue global loads early; latency overlaps TDM wait/barrier)
    const int kA = K + lh * 16;
    v4f a0 = *(const v4f*)(xrow + kA);
    v4f a1 = *(const v4f*)(xrow + kA + 4);
    v4f a2 = *(const v4f*)(xrow + kA + 8);
    v4f a3 = *(const v4f*)(xrow + kA + 12);
    float af[16];
#pragma unroll
    for (int i = 0; i < 4; ++i) {
      af[i]      = a0[i];
      af[4 + i]  = a1[i];
      af[8 + i]  = a2[i];
      af[12 + i] = a3[i];
    }
    v16bf ahi, alo;
#pragma unroll
    for (int i = 0; i < 16; ++i) {
      __bf16 h = (__bf16)af[i];
      ahi[i] = h;
      alo[i] = (__bf16)(af[i] - (float)h);
    }

    // --- TDM: issue next strip, wait for current strip (wave 0 owns TENSORcnt)
    if (wv == 0) {
      if (k + 1 < NKS) {
        tdm_stage_B(Wsplit + (size_t)(k + 1) * KSTEP, &ldsB[nxt][0], lane);
        __builtin_amdgcn_s_wait_tensorcnt((short)1);
      } else {
        __builtin_amdgcn_s_wait_tensorcnt((short)0);
      }
    }
    __syncthreads();   // current buffer visible to all waves

    // --- B tiles from LDS (80 B row pitch -> conflict-free), 12 WMMAs
#pragma unroll
    for (int t = 0; t < 4; ++t) {
      const int n = t * 16 + lm;           // B column owned by this lane
      const v8bf* ph = (const v8bf*)&ldsB[cur][n * BROW + lh * 16];
      const v8bf* pl = (const v8bf*)&ldsB[cur][(64 + n) * BROW + lh * 16];
      v8bf h0 = ph[0], h1 = ph[1];
      v8bf l0 = pl[0], l1 = pl[1];
      v16bf bhi, blo;
#pragma unroll
      for (int i = 0; i < 8; ++i) {
        bhi[i] = h0[i]; bhi[8 + i] = h1[i];
        blo[i] = l0[i]; blo[8 + i] = l1[i];
      }
      // fp32 ~ hi*hi + lo*hi + hi*lo  (f32 accumulate)
      acc[t] = __builtin_amdgcn_wmma_f32_16x16x32_bf16(
          false, ahi, false, bhi, (short)0, acc[t], false, false);
      acc[t] = __builtin_amdgcn_wmma_f32_16x16x32_bf16(
          false, alo, false, bhi, (short)0, acc[t], false, false);
      acc[t] = __builtin_amdgcn_wmma_f32_16x16x32_bf16(
          false, ahi, false, blo, (short)0, acc[t], false, false);
    }
    __syncthreads();   // all waves done reading 'cur' before it is re-staged
  }

  // --- D layout -> LDS: VGPR v holds M=v (lanes 0-15) / M=v+8 (lanes 16-31).
#pragma unroll
  for (int t = 0; t < 4; ++t)
#pragma unroll
    for (int v = 0; v < 8; ++v)
      lgs[wv][lh ? (v + 8) : v][t * 16 + lm] = acc[t][v];
  __syncthreads();

  // --- Router epilogue: lanes 0-15 of each wave each own one token row.
  if (lane < 16) {
    const float* l = lgs[wv][lane];
    float v1 = -INFINITY, v2 = -INFINITY;
    int i1 = 0, i2 = 0;
#pragma unroll 4
    for (int e = 0; e < N_EXP; ++e) {
      float le = l[e];
      if (le > v1) { v2 = v1; i2 = i1; v1 = le; i1 = e; }
      else if (le > v2) { v2 = le; i2 = e; }
    }
    float e2 = __expf(v2 - v1);
    float p1 = 1.0f / (1.0f + e2);
    int tok = rowBase + lane;
    out[2 * tok]     = p1;
    out[2 * tok + 1] = e2 * p1;
    float* idxOut = out + 2 * tokens;
    idxOut[2 * tok]     = (float)i1;
    idxOut[2 * tok + 1] = (float)i2;

    float denom = 0.0f;
#pragma unroll 4
    for (int e = 0; e < N_EXP; ++e) denom += __expf(l[e] - v1);
    float inv = 1.0f / denom;
#pragma unroll 4
    for (int e = 0; e < N_EXP; ++e)
      atomicAdd(&wgExpert[e], __expf(l[e] - v1) * inv);
  }
  __syncthreads();

  if (tid < N_EXP) atomicAdd(&gExpert[tid], wgExpert[tid]);
}

// ---------------------------------------------------------------------------
// Kernel 3: expert_probs mean + entropy-style load-balance loss.
// ---------------------------------------------------------------------------
__global__ void router_finalize(const float* __restrict__ gExpert,
                                float* __restrict__ out, int tokens) {
  __shared__ float red[N_EXP];
  int t = threadIdx.x;
  float p = gExpert[t] / (float)tokens;
  red[t] = p * __logf(p + 1e-8f);
  __syncthreads();
  for (int s = 32; s > 0; s >>= 1) {
    if (t < s) red[t] += red[t + s];
    __syncthreads();
  }
  if (t == 0) out[4 * tokens] = 0.01f * red[0];
}

// ---------------------------------------------------------------------------
extern "C" void kernel_launch(void* const* d_in, const int* in_sizes, int n_in,
                              void* d_out, int out_size, void* d_ws, size_t ws_size,
                              hipStream_t stream) {
  const float* x = (const float*)d_in[0];
  const float* W = (const float*)d_in[1];
  float* out = (float*)d_out;
  const int tokens = in_sizes[0] / D_MODEL;   // 32768

  // Workspace: Wsplit = [hi 64x2048 | lo 64x2048] bf16, then gExpert (64 f32)
  __bf16* Whi = (__bf16*)d_ws;
  __bf16* Wlo = Whi + (size_t)N_EXP * D_MODEL;
  float* gExpert = (float*)(Wlo + (size_t)N_EXP * D_MODEL);

  router_prep<<<(N_EXP * D_MODEL + 255) / 256, 256, 0, stream>>>(W, Whi, Wlo, gExpert);
  router_gemm<<<tokens / 64, 128, 0, stream>>>(x, Whi, gExpert, out, tokens);
  router_finalize<<<1, N_EXP, 0, stream>>>(gExpert, out, tokens);
}
